// SimpleGCN_10926396801662
// MI455X (gfx1250) — matrix operand order repaired and checked by reading, not compiled, not run on hardware
//
#include <hip/hip_runtime.h>

#define N_NODES    100000
#define N_EDGES    1600000
#define NUM_GRAPHS 128
#define IN_DIM     128
#define HID        64
#define HID2       32
#define NCLS       10

#define CDIV(a, b) (((a) + (b) - 1) / (b))

typedef __attribute__((ext_vector_type(16))) __bf16          v16bf;
typedef __attribute__((ext_vector_type(16))) unsigned short  v16us;
typedef __attribute__((ext_vector_type(8)))  float           v8f;

static __device__ __forceinline__ __bf16 f32_to_bf16(float f) {
    // round-to-nearest-even truncation to bf16, pure bit ops
    unsigned int u = __builtin_bit_cast(unsigned int, f);
    u += 0x7FFFu + ((u >> 16) & 1u);
    unsigned short s = (unsigned short)(u >> 16);
    return __builtin_bit_cast(__bf16, s);
}

// ---------------- degree / normalization ----------------

__global__ void k_init_deg(float* deg, int n) {
    int i = blockIdx.x * blockDim.x + threadIdx.x;
    if (i < n) deg[i] = 1.0f;  // self-loop weight
}

__global__ void k_deg_accum(const int* __restrict__ col, const float* __restrict__ ew,
                            float* deg, int E) {
    int e = blockIdx.x * blockDim.x + threadIdx.x;
    if (e < E) atomicAdd(&deg[col[e]], ew[e]);
}

__global__ void k_dinv(const float* __restrict__ deg, float* dinv, int n) {
    int i = blockIdx.x * blockDim.x + threadIdx.x;
    if (i < n) {
        float d = deg[i];
        dinv[i] = d > 0.0f ? rsqrtf(d) : 0.0f;
    }
}

__global__ void k_norm(const int* __restrict__ row, const int* __restrict__ col,
                       const float* __restrict__ ew, const float* __restrict__ dinv,
                       float* nrm, int E) {
    int e = blockIdx.x * blockDim.x + threadIdx.x;
    if (e < E) nrm[e] = dinv[row[e]] * ew[e] * dinv[col[e]];
}

// ---------------- WMMA bf16 GEMM:  C[M,64] = A[M,K] @ B[K,64] ----------------
// M % 16 == 0, K % 32 == 0, K <= 128.
// 10 waves / block; each wave owns one 16-row stripe and computes all 4 N-tiles
// (one A fragment feeds 4 WMMAs). B is staged once per block into LDS as bf16
// in WMMA fragment order, so each lane pulls its 32-byte fragment via ds_load.

#define GEMM_WAVES 10

__global__ void __launch_bounds__(32 * GEMM_WAVES)
k_wmma_gemm64(const float* __restrict__ A, const float* __restrict__ B,
              float* __restrict__ C, int tilesM, int K) {
    __shared__ unsigned short ldsB[(128 / 32) * 4 * 512];   // 16 KB max (K=128)

    const int tid    = threadIdx.x;
    const int lane   = tid & 31;
    const int wave   = tid >> 5;
    const int nsteps = K >> 5;

    // ---- stage B -> LDS (bf16, fragment order: [s][nt][lane][i]) ----
    const int totalB = nsteps * 4 * 512;
    for (int idx = tid; idx < totalB; idx += 32 * GEMM_WAVES) {
        int fi  = idx >> 9;           // fragment = s*4 + nt
        int rem = idx & 511;
        int l   = rem >> 4;           // lane slot
        int i   = rem & 15;           // element within fragment
        int s   = fi >> 2;
        int nt  = fi & 3;
        int n   = (nt << 4) + (l & 15);
        int kb  = (s << 5) + ((l >> 4) << 3) + (i & 7) + ((i >> 3) << 4);    // K = s*32 + 8*half + (i&7) + 16*(i>=8)
        ldsB[idx] = __builtin_bit_cast(unsigned short, f32_to_bf16(B[kb * 64 + n]));
    }
    __syncthreads();

    const int tileM = blockIdx.x * GEMM_WAVES + wave;
    if (tileM < tilesM) {             // wave-uniform; EXEC all-ones inside
        const int half = lane >> 4;
        const int l16  = lane & 15;
        const float* arow = A + (size_t)(tileM * 16 + l16) * K;

        v8f acc0 = {}, acc1 = {}, acc2 = {}, acc3 = {};
        for (int s = 0; s < nsteps; ++s) {
            // A fragment: elem 0..7 <-> K = s*32 + 8*half + i; elem 8..15 <-> +16
            const float* ap = arow + (s << 5) + (half << 3);
            float4 a0 = *(const float4*)(ap + 0);
            float4 a1 = *(const float4*)(ap + 4);
            float4 a2 = *(const float4*)(ap + 16);
            float4 a3 = *(const float4*)(ap + 20);
            v16bf a;
            a[0]  = f32_to_bf16(a0.x); a[1]  = f32_to_bf16(a0.y);
            a[2]  = f32_to_bf16(a0.z); a[3]  = f32_to_bf16(a0.w);
            a[4]  = f32_to_bf16(a1.x); a[5]  = f32_to_bf16(a1.y);
            a[6]  = f32_to_bf16(a1.z); a[7]  = f32_to_bf16(a1.w);
            a[8]  = f32_to_bf16(a2.x); a[9]  = f32_to_bf16(a2.y);
            a[10] = f32_to_bf16(a2.z); a[11] = f32_to_bf16(a2.w);
            a[12] = f32_to_bf16(a3.x); a[13] = f32_to_bf16(a3.y);
            a[14] = f32_to_bf16(a3.z); a[15] = f32_to_bf16(a3.w);

            const unsigned short* bbase = ldsB + ((s << 2) << 9) + (lane << 4);
            v16bf b0 = __builtin_bit_cast(v16bf, *(const v16us*)(bbase + 0 * 512));
            v16bf b1 = __builtin_bit_cast(v16bf, *(const v16us*)(bbase + 1 * 512));
            v16bf b2 = __builtin_bit_cast(v16bf, *(const v16us*)(bbase + 2 * 512));
            v16bf b3 = __builtin_bit_cast(v16bf, *(const v16us*)(bbase + 3 * 512));

            acc0 = __builtin_amdgcn_wmma_f32_16x16x32_bf16(false, a, false, b0, (short)0, acc0, false, false);
            acc1 = __builtin_amdgcn_wmma_f32_16x16x32_bf16(false, a, false, b1, (short)0, acc1, false, false);
            acc2 = __builtin_amdgcn_wmma_f32_16x16x32_bf16(false, a, false, b2, (short)0, acc2, false, false);
            acc3 = __builtin_amdgcn_wmma_f32_16x16x32_bf16(false, a, false, b3, (short)0, acc3, false, false);
        }

        // D element v -> row v + 8*half, col nt*16 + l16
        float* crow = C + (size_t)(tileM * 16 + (half << 3)) * 64 + l16;
#pragma unroll
        for (int v = 0; v < 8; ++v) {
            crow[(size_t)v * 64 +  0] = acc0[v];
            crow[(size_t)v * 64 + 16] = acc1[v];
            crow[(size_t)v * 64 + 32] = acc2[v];
            crow[(size_t)v * 64 + 48] = acc3[v];
        }
    }
}

// ---------------- edge scatter: out[c] += norm * h[r] ----------------
// one thread per (edge, 4-feature chunk): float4 gather + 4 f32 atomics

__global__ void k_scatter(const int* __restrict__ row, const int* __restrict__ col,
                          const float* __restrict__ nrm, const float* __restrict__ h,
                          float* out, int E) {
    long idx = (long)blockIdx.x * blockDim.x + threadIdx.x;
    if (idx >= (long)E * (HID / 4)) return;
    int e  = (int)(idx >> 4);                      // HID/4 == 16
    int f4 = (int)(idx & 15);
    int r = row[e], c = col[e];
    float w = nrm[e];
    float4 hv = ((const float4*)(h + (size_t)r * HID))[f4];
    float* o = out + (size_t)c * HID + f4 * 4;
    atomicAdd(o + 0, w * hv.x);
    atomicAdd(o + 1, w * hv.y);
    atomicAdd(o + 2, w * hv.z);
    atomicAdd(o + 3, w * hv.w);
}

// finalize: out = relu(out + dinv[i]^2 * h + bias)   (self-loop term + bias + relu)
__global__ void k_finalize(float* out, const float* __restrict__ h,
                           const float* __restrict__ dinv, const float* __restrict__ bias,
                           int n) {
    int idx = blockIdx.x * blockDim.x + threadIdx.x;
    if (idx < n * HID) {
        int i = idx >> 6;          // /HID
        int f = idx & (HID - 1);
        float di = dinv[i];
        float v = out[idx] + di * di * h[idx] + bias[f];
        out[idx] = v > 0.0f ? v : 0.0f;
    }
}

// ---------------- global mean pool ----------------

__global__ void k_pool_sum(const float* __restrict__ h, const int* __restrict__ batch,
                           float* sums, int n) {
    int idx = blockIdx.x * blockDim.x + threadIdx.x;
    if (idx < n * HID) {
        int i = idx >> 6;
        int f = idx & (HID - 1);
        atomicAdd(&sums[(size_t)batch[i] * HID + f], h[idx]);
    }
}

__global__ void k_pool_cnt(const int* __restrict__ batch, float* cnts, int n) {
    int i = blockIdx.x * blockDim.x + threadIdx.x;
    if (i < n) atomicAdd(&cnts[batch[i]], 1.0f);
}

__global__ void k_pool_div(float* sums, const float* __restrict__ cnts, int total) {
    int idx = blockIdx.x * blockDim.x + threadIdx.x;
    if (idx < total) {
        int g = idx >> 6;
        float c = cnts[g];
        if (c < 1.0f) c = 1.0f;
        sums[idx] /= c;
    }
}

// ---------------- tiny classifier head (scalar; negligible work) ----------------

__global__ void k_fc1(const float* __restrict__ pooled, const float* __restrict__ W,
                      const float* __restrict__ b, float* z) {
    int idx = blockIdx.x * blockDim.x + threadIdx.x;   // G * HID2
    if (idx < NUM_GRAPHS * HID2) {
        int g = idx / HID2, j = idx % HID2;
        float s = b[j];
        for (int k = 0; k < HID; ++k) s += pooled[g * HID + k] * W[k * HID2 + j];
        z[idx] = s > 0.0f ? s : 0.0f;
    }
}

__global__ void k_fc2(const float* __restrict__ z, const float* __restrict__ W,
                      const float* __restrict__ b, float* out) {
    int idx = blockIdx.x * blockDim.x + threadIdx.x;   // G * NCLS
    if (idx < NUM_GRAPHS * NCLS) {
        int g = idx / NCLS, c = idx % NCLS;
        float s = b[c];
        for (int k = 0; k < HID2; ++k) s += z[g * HID2 + k] * W[k * NCLS + c];
        out[idx] = s;
    }
}

// ---------------- launcher ----------------

extern "C" void kernel_launch(void* const* d_in, const int* in_sizes, int n_in,
                              void* d_out, int out_size, void* d_ws, size_t ws_size,
                              hipStream_t stream) {
    (void)in_sizes; (void)n_in; (void)out_size; (void)ws_size;

    const float* x   = (const float*)d_in[0];
    const int*   ei  = (const int*)d_in[1];
    const float* ew  = (const float*)d_in[2];
    const int*   bat = (const int*)d_in[3];
    const float* W1  = (const float*)d_in[4];
    const float* b1  = (const float*)d_in[5];
    const float* W2  = (const float*)d_in[6];
    const float* b2  = (const float*)d_in[7];
    const float* Wc1 = (const float*)d_in[8];
    const float* bc1 = (const float*)d_in[9];
    const float* Wc2 = (const float*)d_in[10];
    const float* bc2 = (const float*)d_in[11];
    float* outp = (float*)d_out;

    const int* row = ei;             // edge_index[0]
    const int* col = ei + N_EDGES;   // edge_index[1]

    char*  ws  = (char*)d_ws;
    size_t off = 0;
    auto alloc = [&](size_t bytes) -> void* {
        void* p = ws + off;
        off += (bytes + 255) & ~(size_t)255;
        return p;
    };
    float* deg  = (float*)alloc((size_t)N_NODES * 4);
    float* dinv = (float*)alloc((size_t)N_NODES * 4);
    float* nrm  = (float*)alloc((size_t)N_EDGES * 4);
    float* bufA = (float*)alloc((size_t)N_NODES * HID * 4);
    float* bufB = (float*)alloc((size_t)N_NODES * HID * 4);
    float* sums = (float*)alloc((size_t)NUM_GRAPHS * HID * 4);
    float* cnts = (float*)alloc((size_t)NUM_GRAPHS * 4);
    float* zbuf = (float*)alloc((size_t)NUM_GRAPHS * HID2 * 4);

    const int B256 = 256;
    const int tilesM = N_NODES / 16;                       // 6250 (exact)
    const int gemmBlocks = CDIV(tilesM, GEMM_WAVES);       // 625
    const int gemmThreads = 32 * GEMM_WAVES;               // 320

    // --- normalization coefficients ---
    k_init_deg<<<CDIV(N_NODES, B256), B256, 0, stream>>>(deg, N_NODES);
    k_deg_accum<<<CDIV(N_EDGES, B256), B256, 0, stream>>>(col, ew, deg, N_EDGES);
    k_dinv<<<CDIV(N_NODES, B256), B256, 0, stream>>>(deg, dinv, N_NODES);
    k_norm<<<CDIV(N_EDGES, B256), B256, 0, stream>>>(row, col, ew, dinv, nrm, N_EDGES);

    // --- layer 1: h1 = x @ W1 (WMMA) ; aggregate ; +b1, relu ---
    k_wmma_gemm64<<<gemmBlocks, gemmThreads, 0, stream>>>(x, W1, bufA, tilesM, IN_DIM);
    hipMemsetAsync(bufB, 0, (size_t)N_NODES * HID * 4, stream);
    {
        long work = (long)N_EDGES * (HID / 4);
        k_scatter<<<(int)CDIV(work, B256), B256, 0, stream>>>(row, col, nrm, bufA, bufB, N_EDGES);
    }
    k_finalize<<<CDIV(N_NODES * HID, B256), B256, 0, stream>>>(bufB, bufA, dinv, b1, N_NODES);

    // --- layer 2: h2 = h1 @ W2 (WMMA) ; aggregate ; +b2, relu ---
    k_wmma_gemm64<<<gemmBlocks, gemmThreads, 0, stream>>>(bufB, W2, bufA, tilesM, HID);
    hipMemsetAsync(bufB, 0, (size_t)N_NODES * HID * 4, stream);   // stream-ordered after GEMM2
    {
        long work = (long)N_EDGES * (HID / 4);
        k_scatter<<<(int)CDIV(work, B256), B256, 0, stream>>>(row, col, nrm, bufA, bufB, N_EDGES);
    }
    k_finalize<<<CDIV(N_NODES * HID, B256), B256, 0, stream>>>(bufB, bufA, dinv, b2, N_NODES);

    // --- global mean pool ---
    hipMemsetAsync(sums, 0, (size_t)NUM_GRAPHS * HID * 4, stream);
    hipMemsetAsync(cnts, 0, (size_t)NUM_GRAPHS * 4, stream);
    k_pool_sum<<<CDIV(N_NODES * HID, B256), B256, 0, stream>>>(bufB, bat, sums, N_NODES);
    k_pool_cnt<<<CDIV(N_NODES, B256), B256, 0, stream>>>(bat, cnts, N_NODES);
    k_pool_div<<<CDIV(NUM_GRAPHS * HID, B256), B256, 0, stream>>>(sums, cnts, NUM_GRAPHS * HID);

    // --- classifier head ---
    k_fc1<<<CDIV(NUM_GRAPHS * HID2, B256), B256, 0, stream>>>(sums, Wc1, bc1, zbuf);
    k_fc2<<<CDIV(NUM_GRAPHS * NCLS, B256), B256, 0, stream>>>(zbuf, Wc2, bc2, outp);
}
